// Encoder_53274774340258
// MI455X (gfx1250) — matrix operand order repaired and checked by reading, not compiled, not run on hardware
//
#include <hip/hip_runtime.h>

// ---------------------------------------------------------------------------
// Types for CDNA5 WMMA (wave32): bf16 A/B operands, f32 accumulators.
// ---------------------------------------------------------------------------
typedef __attribute__((ext_vector_type(16))) __bf16 v16bf;
typedef __attribute__((ext_vector_type(8)))  float  v8f;

union V16BF {
    v16bf v;
    uint4 u[2];
    unsigned u32[8];
    __bf16 e[16];
};

// Problem dims
constexpr int Bz = 4, Pz = 2048, Dz = 768, Hz = 12, HDz = 64;
constexpr int BP = Bz * Pz;          // 8192
constexpr int D3 = 3 * Dz;           // 2304
constexpr int D4 = 4 * Dz;           // 3072
constexpr float SCALE = 0.125f;      // HD^-0.5 = 1/8

// A-operand (16-bit, 16xK) VGPR layout (contiguous, matches B fills):
// lane half lh in {0,1}; element e in [0,16):  K = e + 16*lh
__device__ __forceinline__ int a_k_of(int e, int lh) {
    return e + (lh << 4);
}

// ---------------------------------------------------------------------------
// fp32 -> bf16 weight cast
// ---------------------------------------------------------------------------
__global__ void cast_w_kernel(const float* __restrict__ src, __bf16* __restrict__ dst, int n) {
    int i = blockIdx.x * blockDim.x + threadIdx.x;
    if (i < n) dst[i] = (__bf16)src[i];
}

// ---------------------------------------------------------------------------
// LayerNorm over D=768 + cast to bf16.  One block (256 thr) per row.
// ---------------------------------------------------------------------------
__global__ __launch_bounds__(256)
void ln_cast_kernel(const float* __restrict__ x, const float* __restrict__ g,
                    const float* __restrict__ bta, __bf16* __restrict__ out) {
    __shared__ float rs[256], rq[256];
    const int row = blockIdx.x, tid = threadIdx.x;
    const float* xr = x + (size_t)row * Dz;
    float v0 = xr[tid], v1 = xr[tid + 256], v2 = xr[tid + 512];
    rs[tid] = v0 + v1 + v2;
    rq[tid] = v0 * v0 + v1 * v1 + v2 * v2;
    __syncthreads();
    for (int s = 128; s > 0; s >>= 1) {
        if (tid < s) { rs[tid] += rs[tid + s]; rq[tid] += rq[tid + s]; }
        __syncthreads();
    }
    const float mu   = rs[0] * (1.0f / Dz);
    const float var  = rq[0] * (1.0f / Dz) - mu * mu;
    const float rstd = rsqrtf(var + 1e-5f);
    __bf16* orow = out + (size_t)row * Dz;
    orow[tid]       = (__bf16)((v0 - mu) * rstd * g[tid]       + bta[tid]);
    orow[tid + 256] = (__bf16)((v1 - mu) * rstd * g[tid + 256] + bta[tid + 256]);
    orow[tid + 512] = (__bf16)((v2 - mu) * rstd * g[tid + 512] + bta[tid + 512]);
}

// ---------------------------------------------------------------------------
// Tiled bf16 WMMA GEMM: C(MxN) = A(MxK) * B(KxN), fused epilogues.
// Block tile 128x64, 8 waves (256 thr), wave w owns rows [16w,16w+16), N=64.
// MODE 0: +bias, q/k head-major scatter (q scaled by 1/8), V TRANSPOSED scatter
// MODE 1: +bias, exact GELU, bf16 out
// MODE 2: +bias, +residual, f32 out
// ---------------------------------------------------------------------------
constexpr int AST = 40;  // LDS A row stride (bf16 elems): 80B rows, 16B aligned
constexpr int BST = 40;  // LDS B^T col stride

template <int MODE>
__global__ __launch_bounds__(256)
void gemm_bf16_kernel(const __bf16* __restrict__ A, const __bf16* __restrict__ Bm,
                      int M, int N, int K,
                      const float* __restrict__ bias,
                      float* __restrict__ outF, __bf16* __restrict__ outH,
                      __bf16* __restrict__ qO, __bf16* __restrict__ kO, __bf16* __restrict__ vO,
                      const float* __restrict__ resid) {
    __shared__ __bf16 As[128 * AST];   // A tile 128x32, row-major
    __shared__ __bf16 Bt[64 * BST];    // B tile 32x64, stored transposed: Bt[n][k]

    const int tid  = threadIdx.x;
    const int w    = tid >> 5;
    const int lane = tid & 31;
    const int lm   = lane & 15;
    const int lh   = lane >> 4;
    const int n0   = blockIdx.x * 64;
    const int m0   = blockIdx.y * 128;

    v8f acc[4] = {v8f{}, v8f{}, v8f{}, v8f{}};

    for (int k0 = 0; k0 < K; k0 += 32) {
        // ---- stage A: 128x32 bf16, 16 elems/thread (32B contiguous) ----
        {
            const int row = tid >> 1, koff = (tid & 1) * 16;
            const uint4* src = (const uint4*)(A + (size_t)(m0 + row) * K + k0 + koff);
            uint4 a0 = src[0], a1 = src[1];
            if (k0 + 32 < K)  // prefetch next A tile (global_prefetch_b8)
                __builtin_prefetch(A + (size_t)(m0 + row) * K + k0 + 32 + koff, 0, 1);
            uint4* dst = (uint4*)&As[row * AST + koff];
            dst[0] = a0; dst[1] = a1;
        }
        // ---- stage B transposed: 32x64 bf16, 8 elems/thread ----
        {
            const int row = tid >> 3, coff = (tid & 7) * 8;
            const __bf16* src = Bm + (size_t)(k0 + row) * N + n0 + coff;
            if (k0 + 32 < K)
                __builtin_prefetch(src + (size_t)32 * N, 0, 1);
#pragma unroll
            for (int j = 0; j < 8; ++j) Bt[(coff + j) * BST + row] = src[j];
        }
        __syncthreads();

        // ---- A operand (16x32 bf16) in contiguous k order (k = e + 16*lh), same as B ----
        V16BF a;
#pragma unroll
        for (int p = 0; p < 8; ++p) {
            const int kk = a_k_of(2 * p, lh);   // even -> dword aligned pair
            a.u32[p] = *(const unsigned*)&As[(16 * w + lm) * AST + kk];
        }
        // ---- 4 B operands (32x16) + WMMA ----
#pragma unroll
        for (int nt = 0; nt < 4; ++nt) {
            V16BF b;
            const uint4* bp = (const uint4*)&Bt[(nt * 16 + lm) * BST + 16 * lh];
            b.u[0] = bp[0]; b.u[1] = bp[1];
            acc[nt] = __builtin_amdgcn_wmma_f32_16x16x32_bf16(
                false, a.v, false, b.v, (short)0, acc[nt], false, false);
        }
        __syncthreads();
    }

    // ---- epilogue: C/D layout => M = r + 8*lh, N = lm ----
#pragma unroll
    for (int nt = 0; nt < 4; ++nt) {
#pragma unroll
        for (int r = 0; r < 8; ++r) {
            const int row = m0 + 16 * w + r + 8 * lh;
            const int col = n0 + nt * 16 + lm;
            float vv = acc[nt][r] + bias[col];
            if constexpr (MODE == 0) {
                const int which = col / Dz;            // 0=q 1=k 2=v
                const int rem   = col - which * Dz;
                const int hh    = rem >> 6;
                const int hd    = rem & 63;
                const int bb    = row >> 11;
                const int pp    = row & 2047;
                if (which == 0) vv *= SCALE;           // fold 1/sqrt(HD) into q
                if (which == 2) {
                    // V stored TRANSPOSED per head: [B,H,HD,P] so the PV-gemm
                    // B-operand is contiguous along the q-position (K) axis.
                    vO[((size_t)(bb * Hz + hh) * HDz + hd) * Pz + pp] = (__bf16)vv;
                } else {
                    __bf16* dst = (which == 0) ? qO : kO;
                    dst[(((size_t)(bb * Hz + hh) * Pz) + pp) * HDz + hd] = (__bf16)vv;
                }
            } else if constexpr (MODE == 1) {
                vv = 0.5f * vv * (1.0f + erff(vv * 0.70710678118654752f));  // exact GELU
                outH[(size_t)row * N + col] = (__bf16)vv;
            } else {
                outF[(size_t)row * N + col] = vv + resid[(size_t)row * N + col];
            }
        }
    }
}

// ---------------------------------------------------------------------------
// Fused attention with softmax over the HEADS axis.
// Grid: B * (P/16) workgroups; 12 waves = 12 heads per workgroup.
// Each WG owns a 16-row p-tile, streams 32-wide q-tiles:
//   S[h] = q_tile @ k_tile^T  (WMMA, K=64 in two bf16 slabs)  -> LDS (f32)
//   cross-head softmax over the 12 S values per (p,q)          -> LDS (bf16)
//   out[h] += attn_tile @ v_tile (WMMA, K=32; V pre-transposed)
// S is never materialized in HBM (saves ~1.6 GB of traffic); K/V stay in L2.
// ---------------------------------------------------------------------------
__global__ __launch_bounds__(384)
void attn_kernel(const __bf16* __restrict__ q, const __bf16* __restrict__ k,
                 const __bf16* __restrict__ vt, const float* __restrict__ x,
                 float* __restrict__ x2) {
    __shared__ float  Sld[Hz][16][32];   // 24 KB logits (then exp values)
    __shared__ __bf16 Ald[Hz][16][32];   // 12 KB normalized attn

    const int tid  = threadIdx.x;
    const int h    = tid >> 5;           // wave id == head
    const int lane = tid & 31;
    const int lm   = lane & 15;
    const int lh   = lane >> 4;
    const int b    = blockIdx.x >> 7;
    const int p0   = (blockIdx.x & 127) << 4;

    const __bf16* qh = q  + ((size_t)(b * Hz + h) * Pz) * HDz;
    const __bf16* kh = k  + ((size_t)(b * Hz + h) * Pz) * HDz;
    const __bf16* vh = vt + ((size_t)(b * Hz + h) * HDz) * Pz;   // [HD][P]

    // q A-operands for both 32-wide K slabs of HD=64 (persistent across the q loop)
    V16BF aq[2];
#pragma unroll
    for (int s = 0; s < 2; ++s)
#pragma unroll
        for (int p = 0; p < 8; ++p) {
            const int hd = a_k_of(2 * p, lh) + 32 * s;
            aq[s].u32[p] = *(const unsigned*)(qh + (size_t)(p0 + lm) * HDz + hd);
        }

    v8f acc[4] = {v8f{}, v8f{}, v8f{}, v8f{}};

    for (int q0 = 0; q0 < Pz; q0 += 32) {
        // prefetch next iteration's K rows and V columns
        if (q0 + 32 < Pz) {
            __builtin_prefetch(kh + (size_t)(q0 + 32 + lm) * HDz + 16 * lh, 0, 1);
            __builtin_prefetch(vh + (size_t)lm * Pz + q0 + 32 + 16 * lh, 0, 1);
        }
        // ---- logits: S = (q*scale) @ k^T for this head's 16x32 tile ----
#pragma unroll
        for (int nt = 0; nt < 2; ++nt) {
            v8f c = {};
#pragma unroll
            for (int s = 0; s < 2; ++s) {
                V16BF bk;  // B operand: K=hd-slab, N=key position; rows of k are contiguous
                const uint4* bp =
                    (const uint4*)(kh + (size_t)(q0 + 16 * nt + lm) * HDz + 16 * lh + 32 * s);
                bk.u[0] = bp[0]; bk.u[1] = bp[1];
                c = __builtin_amdgcn_wmma_f32_16x16x32_bf16(
                    false, aq[s].v, false, bk.v, (short)0, c, false, false);
            }
#pragma unroll
            for (int r = 0; r < 8; ++r) Sld[h][r + 8 * lh][16 * nt + lm] = c[r];
        }
        __syncthreads();

        // ---- softmax across the 12 heads for each of the 512 (p,q) cells ----
        for (int idx = tid; idx < 512; idx += 384) {
            const int pp = idx >> 5, qq = idx & 31;
            float mx = -1e30f;
#pragma unroll
            for (int hh = 0; hh < Hz; ++hh) mx = fmaxf(mx, Sld[hh][pp][qq]);
            float sum = 0.0f;
#pragma unroll
            for (int hh = 0; hh < Hz; ++hh) {
                const float e = __expf(Sld[hh][pp][qq] - mx);
                Sld[hh][pp][qq] = e;   // cache exp: avoids a second transcendental pass
                sum += e;
            }
            const float inv = 1.0f / sum;
#pragma unroll
            for (int hh = 0; hh < Hz; ++hh)
                Ald[hh][pp][qq] = (__bf16)(Sld[hh][pp][qq] * inv);
        }
        __syncthreads();

        // ---- out += attn @ v  (A: 16x32 attn tile from LDS; B: V^T contiguous) ----
        V16BF aa;
#pragma unroll
        for (int p = 0; p < 8; ++p) {
            const int kk = a_k_of(2 * p, lh);
            aa.u32[p] = *(const unsigned*)&Ald[h][lm][kk];
        }
#pragma unroll
        for (int nt = 0; nt < 4; ++nt) {
            V16BF bv;  // element e: K = e + 16*lh (q pos), N = lm (hd col) -> contiguous in e
            const uint4* bp = (const uint4*)(vh + (size_t)(nt * 16 + lm) * Pz + q0 + 16 * lh);
            bv.u[0] = bp[0]; bv.u[1] = bp[1];
            acc[nt] = __builtin_amdgcn_wmma_f32_16x16x32_bf16(
                false, aa.v, false, bv.v, (short)0, acc[nt], false, false);
        }
        __syncthreads();  // protect LDS tiles before next iteration rewrites them
    }

    // ---- epilogue: head-major -> (B,P,D), add residual x ----
#pragma unroll
    for (int nt = 0; nt < 4; ++nt)
#pragma unroll
        for (int r = 0; r < 8; ++r) {
            const int row = p0 + r + 8 * lh;
            const int col = h * HDz + nt * 16 + lm;
            const size_t idx = ((size_t)b * Pz + row) * Dz + col;
            x2[idx] = acc[nt][r] + x[idx];
        }
}

// ---------------------------------------------------------------------------
// Launcher
// ---------------------------------------------------------------------------
extern "C" void kernel_launch(void* const* d_in, const int* in_sizes, int n_in,
                              void* d_out, int out_size, void* d_ws, size_t ws_size,
                              hipStream_t stream) {
    const float* x     = (const float*)d_in[0];
    const float* ln1w  = (const float*)d_in[1];
    const float* ln1b  = (const float*)d_in[2];
    const float* wqkv  = (const float*)d_in[3];
    const float* bqkv  = (const float*)d_in[4];
    const float* ln2w  = (const float*)d_in[5];
    const float* ln2b  = (const float*)d_in[6];
    const float* wfc1  = (const float*)d_in[7];
    const float* bfc1  = (const float*)d_in[8];
    const float* wfc2  = (const float*)d_in[9];
    const float* bfc2  = (const float*)d_in[10];
    float* out = (float*)d_out;

    // Workspace carve-up (all offsets 256B-aligned by construction)
    char* ws = (char*)d_ws;
    size_t off = 0;
    auto carve = [&](size_t bytes) { char* p = ws + off; off += (bytes + 255) & ~(size_t)255; return p; };
    __bf16* h_bf    = (__bf16*)carve((size_t)BP * Dz * 2);       // ln1(x), bf16
    __bf16* wqkv_bf = (__bf16*)carve((size_t)Dz * D3 * 2);
    __bf16* wfc1_bf = (__bf16*)carve((size_t)Dz * D4 * 2);
    __bf16* wfc2_bf = (__bf16*)carve((size_t)D4 * Dz * 2);
    __bf16* q_bf    = (__bf16*)carve((size_t)Bz * Hz * Pz * HDz * 2);
    __bf16* k_bf    = (__bf16*)carve((size_t)Bz * Hz * Pz * HDz * 2);
    __bf16* vt_bf   = (__bf16*)carve((size_t)Bz * Hz * HDz * Pz * 2);  // V transposed
    float*  x2      = (float*)carve((size_t)BP * Dz * 4);        // attn out + residual
    __bf16* h2_bf   = (__bf16*)carve((size_t)BP * Dz * 2);       // ln2(x2), bf16
    __bf16* h3_bf   = (__bf16*)carve((size_t)BP * D4 * 2);       // gelu(fc1), bf16
    (void)ws_size; (void)in_sizes; (void)n_in; (void)out_size;

    // 1) weight casts fp32 -> bf16
    cast_w_kernel<<<(Dz * D3 + 255) / 256, 256, 0, stream>>>(wqkv, wqkv_bf, Dz * D3);
    cast_w_kernel<<<(Dz * D4 + 255) / 256, 256, 0, stream>>>(wfc1, wfc1_bf, Dz * D4);
    cast_w_kernel<<<(D4 * Dz + 255) / 256, 256, 0, stream>>>(wfc2, wfc2_bf, D4 * Dz);

    // 2) LN1 + cast
    ln_cast_kernel<<<BP, 256, 0, stream>>>(x, ln1w, ln1b, h_bf);

    // 3) QKV GEMM (8192x768x2304) with fused bias + q/k scatter + transposed-V scatter
    gemm_bf16_kernel<0><<<dim3(D3 / 64, BP / 128), 256, 0, stream>>>(
        h_bf, wqkv_bf, BP, D3, Dz, bqkv, nullptr, nullptr, q_bf, k_bf, vt_bf, nullptr);

    // 4) fused attention (softmax over heads) + residual
    attn_kernel<<<Bz * (Pz / 16), 384, 0, stream>>>(q_bf, k_bf, vt_bf, x, x2);

    // 5) LN2 + cast
    ln_cast_kernel<<<BP, 256, 0, stream>>>(x2, ln2w, ln2b, h2_bf);

    // 6) FC1 GEMM (8192x768x3072) + bias + exact GELU -> bf16
    gemm_bf16_kernel<1><<<dim3(D4 / 64, BP / 128), 256, 0, stream>>>(
        h2_bf, wfc1_bf, BP, D4, Dz, bfc1, nullptr, h3_bf, nullptr, nullptr, nullptr, nullptr);

    // 7) FC2 GEMM (8192x3072x768) + bias + residual -> f32 output
    gemm_bf16_kernel<2><<<dim3(Dz / 64, BP / 128), 256, 0, stream>>>(
        h3_bf, wfc2_bf, BP, Dz, D4, bfc2, out, nullptr, nullptr, nullptr, nullptr, x2);
}
